// Attention_74560632258875
// MI455X (gfx1250) — matrix operand order repaired and checked by reading, not compiled, not run on hardware
//
#include <hip/hip_runtime.h>
#include <hip/hip_bf16.h>
#include <stdint.h>

// ---------------------------------------------------------------------------
// Types / helpers for gfx1250 WMMA (wave32)
// ---------------------------------------------------------------------------
typedef __bf16 bf16;
typedef bf16  v16bf __attribute__((ext_vector_type(16)));
typedef bf16  v8bf  __attribute__((ext_vector_type(8)));
typedef float v8f   __attribute__((ext_vector_type(8)));

#define LOG2E 1.4426950408889634f

static constexpr int Tn   = 2048;
static constexpr int HIDn = 2048;
static constexpr int NQn  = 16;
static constexpr int NKVn = 8;
static constexpr int Dn   = 128;

__device__ __forceinline__ v8f zero8() {
  v8f z;
#pragma unroll
  for (int i = 0; i < 8; ++i) z[i] = 0.0f;
  return z;
}

// A-fragment (16x32 bf16): lane holds row = lane%16, K = 2*kb..2*kb+15
// (kb = (lane/16)*8).  Two 16B LDS loads from a row pointer.
__device__ __forceinline__ v16bf frag_a(const bf16* row, int kb) {
  v8bf lo = *(const v8bf*)(row + 2 * kb);
  v8bf hi = *(const v8bf*)(row + 2 * kb + 8);
  v16bf r;
#pragma unroll
  for (int i = 0; i < 8; ++i) { r[i] = lo[i]; r[i + 8] = hi[i]; }
  return r;
}

// B-fragment (32x16 bf16): lane holds col = lane%16, 16 consecutive K values
// starting at (lane/16)*16.  One contiguous 32B LDS read.
__device__ __forceinline__ v16bf frag_b(const bf16* p) {
  v8bf lo = *(const v8bf*)(p);
  v8bf hi = *(const v8bf*)(p + 8);
  v16bf r;
#pragma unroll
  for (int i = 0; i < 8; ++i) { r[i] = lo[i]; r[i + 8] = hi[i]; }
  return r;
}

__device__ __forceinline__ v8f wmma_bf16(v16bf a, v16bf b, v8f c) {
  return __builtin_amdgcn_wmma_f32_16x16x32_bf16(false, a, false, b, (short)0, c,
                                                 false, false);
}

// ---------------------------------------------------------------------------
// fp32 -> bf16 convert, 8 elements/thread (2x b128 load -> 1x b128 store)
// ---------------------------------------------------------------------------
__global__ __launch_bounds__(256) void k_cvt_bf16(bf16* __restrict__ o,
                                                  const float* __restrict__ in,
                                                  int n) {
  const int stride = gridDim.x * blockDim.x * 8;
  for (int i = (blockIdx.x * blockDim.x + threadIdx.x) * 8; i < n; i += stride) {
    float4 a = *(const float4*)(in + i);
    float4 b = *(const float4*)(in + i + 4);
    union { uint4 v; bf16 h[8]; } u;
    u.h[0] = (bf16)a.x; u.h[1] = (bf16)a.y; u.h[2] = (bf16)a.z; u.h[3] = (bf16)a.w;
    u.h[4] = (bf16)b.x; u.h[5] = (bf16)b.y; u.h[6] = (bf16)b.z; u.h[7] = (bf16)b.w;
    *(uint4*)(o + i) = u.v;
  }
}

// ---------------------------------------------------------------------------
// Tiled bf16 WMMA GEMM, compile-time shape: C[M,N](f32) = A[M,K] * B[K,N].
// BM=BN=128, BK=32.  128 threads = 4 waves arranged 2(M) x 2(N); each wave
// computes a 64x64 tile = 4x4 WMMA accumulators (8 LDS frag loads : 16 WMMA).
// Double-buffered LDS, register-staged global loads, 1 barrier per K-step,
// prefetch 2 tiles ahead.
// ---------------------------------------------------------------------------
#define GBM 128
#define GBN 128
#define GBK 32
#define GSTR 48  // padded LDS row stride (96B, 16B aligned)

template <int M, int N, int K>
__global__ __launch_bounds__(128) void k_gemm_bf16(
    float* __restrict__ C, const bf16* __restrict__ A,
    const bf16* __restrict__ B) {
  __shared__ bf16 As[2][GBM * GSTR];  // A tile, row-major [128][32]
  __shared__ bf16 Bt[2][GBN * GSTR];  // B tile transposed [n][k] = [128][32]

  const int tid  = threadIdx.x;
  const int lane = tid & 31;
  const int wid  = tid >> 5;
  const int wm   = wid & 1;   // 0..1 -> 64-row slice
  const int wn   = wid >> 1;  // 0..1 -> 64-col slice
  const int half = lane >> 4;
  const int lc   = lane & 15;
  const int m0 = blockIdx.y * GBM;
  const int n0 = blockIdx.x * GBN;

  // per-thread staging mapping
  const int arow = tid >> 2;            // A: 0..31 (+32 per chunk)
  const int acol = (tid & 3) << 3;      // A: col 0/8/16/24
  const int bk2  = (tid >> 4) << 1;     // B: even k row 0..14 (+16 per chunk)
  const int bnc  = (tid & 15) << 3;     // B: col 0..120

  v8f acc[4][4];
#pragma unroll
  for (int i = 0; i < 4; ++i)
#pragma unroll
    for (int j = 0; j < 4; ++j) acc[i][j] = zero8();

  uint4 ra[4];     // A: 4 b128 chunks (rows arow + 32c)
  uint4 rb[2][2];  // B: 2 units x 2 consecutive k-rows

  auto load_tiles = [&](int k0) {
#pragma unroll
    for (int c = 0; c < 4; ++c)
      ra[c] = *(const uint4*)(A + (size_t)(m0 + arow + 32 * c) * K + k0 + acol);
#pragma unroll
    for (int c = 0; c < 2; ++c) {
      rb[c][0] = *(const uint4*)(B + (size_t)(k0 + bk2 + 16 * c) * N + n0 + bnc);
      rb[c][1] =
          *(const uint4*)(B + (size_t)(k0 + bk2 + 16 * c + 1) * N + n0 + bnc);
    }
  };
  auto store_tiles = [&](int buf) {
#pragma unroll
    for (int c = 0; c < 4; ++c)
      *(uint4*)(&As[buf][(arow + 32 * c) * GSTR + acol]) = ra[c];
#pragma unroll
    for (int c = 0; c < 2; ++c) {
      const unsigned short* p0 = (const unsigned short*)&rb[c][0];
      const unsigned short* p1 = (const unsigned short*)&rb[c][1];
#pragma unroll
      for (int i = 0; i < 8; ++i) {
        uint32_t u = (uint32_t)p0[i] | ((uint32_t)p1[i] << 16);
        *(uint32_t*)(&Bt[buf][(bnc + i) * GSTR + bk2 + 16 * c]) = u;
      }
    }
  };

  constexpr int NIT = K / GBK;
  load_tiles(0);
  store_tiles(0);

  for (int it = 0; it < NIT; ++it) {
    __syncthreads();
    const int cur = it & 1;
    const bool more = (it + 1) < NIT;
    if (more) load_tiles((it + 1) * GBK);
    if (it + 2 < NIT) {  // gfx1250 global_prefetch path
      const int k2 = (it + 2) * GBK;
      __builtin_prefetch(A + (size_t)(m0 + arow) * K + k2 + acol, 0, 0);
      __builtin_prefetch(B + (size_t)(k2 + bk2) * N + n0 + bnc, 0, 0);
    }

    const bf16* as = &As[cur][0];
    const bf16* bt = &Bt[cur][0];
    v16bf af[4], bfr[4];
#pragma unroll
    for (int tm = 0; tm < 4; ++tm)
      af[tm] = frag_a(&as[(wm * 64 + tm * 16 + lc) * GSTR], half * 8);
#pragma unroll
    for (int tn = 0; tn < 4; ++tn)
      bfr[tn] = frag_b(&bt[(wn * 64 + tn * 16 + lc) * GSTR + half * 16]);
#pragma unroll
    for (int tm = 0; tm < 4; ++tm)
#pragma unroll
      for (int tn = 0; tn < 4; ++tn)
        acc[tm][tn] = wmma_bf16(af[tm], bfr[tn], acc[tm][tn]);

    if (more) store_tiles(cur ^ 1);
  }

  // C layout: VGPR r -> row (half*8 + r), col = lc.  N is a compile-time
  // constant so the r*N strides become immediate offsets.
#pragma unroll
  for (int tm = 0; tm < 4; ++tm)
#pragma unroll
    for (int tn = 0; tn < 4; ++tn) {
      const size_t base = (size_t)(m0 + wm * 64 + tm * 16 + half * 8) * N +
                          (n0 + wn * 64 + tn * 16 + lc);
#pragma unroll
      for (int r = 0; r < 8; ++r) C[base + (size_t)r * N] = acc[tm][tn][r];
    }
}

// ---------------------------------------------------------------------------
// Fused per-head RMSNorm + RoPE (+ optional softmax scale), f32 -> bf16.
// One 128-thread block per (t, head).  rotate_half with half=64.
// ---------------------------------------------------------------------------
__global__ __launch_bounds__(128) void k_rmsrope(
    bf16* __restrict__ out, const float* __restrict__ in,
    const float* __restrict__ cosT, const float* __restrict__ sinT,
    const float* __restrict__ w, int nheads, float outscale) {
  const int t = blockIdx.x;
  const int h = blockIdx.y;
  const int d = threadIdx.x;
  const int stride = nheads * Dn;

  float x = in[(size_t)t * stride + h * Dn + d];

  float ss = x * x;
#pragma unroll
  for (int off = 1; off < 32; off <<= 1) ss += __shfl_xor(ss, off, 32);
  __shared__ float red[4];
  __shared__ float qn[Dn];
  if ((d & 31) == 0) red[d >> 5] = ss;
  __syncthreads();
  float tot = red[0] + red[1] + red[2] + red[3];
  float rinv = rsqrtf(tot * (1.0f / Dn) + 1e-6f);

  float q = x * rinv * w[d];
  qn[d] = q;
  __syncthreads();
  float rot = (d < 64) ? -qn[d + 64] : qn[d - 64];
  float val = (q * cosT[t * Dn + d] + rot * sinT[t * Dn + d]) * outscale;
  out[(size_t)t * stride + h * Dn + d] = (bf16)val;
}

// ---------------------------------------------------------------------------
// Flash attention (causal, GQA 2:1).  Grid: (T/64, NQ).  128 threads = 4 waves;
// wave w owns query rows q0 + w*16 .. +15.  64-key blocks; S and online softmax
// in registers, P restriped via per-wave LDS for the PV WMMA.
// Q is pre-scaled by D^-0.5.  Output: bf16 [T, NQ*D].
// ---------------------------------------------------------------------------
__global__ __launch_bounds__(128) void k_attn(
    bf16* __restrict__ O, const bf16* __restrict__ Q,
    const bf16* __restrict__ Kp, const bf16* __restrict__ Vp) {
  __shared__ bf16 Qs[64 * 128];    // [qrow][d]
  __shared__ bf16 Ks[64 * 128];    // [key][d]
  __shared__ bf16 Vt[128 * 64];    // transposed: [d][key]
  __shared__ bf16 Ps[4 * 16 * 64]; // per-wave P tile [16][64]

  const int tid  = threadIdx.x;
  const int lane = tid & 31;
  const int wid  = tid >> 5;
  const int half = lane >> 4;
  const int lc   = lane & 15;
  const int h    = blockIdx.y;
  const int q0   = blockIdx.x * 64;
  const int kvh  = h >> 1;  // NQ/NKV = 2
  constexpr int QS = NQn * Dn;   // 2048
  constexpr int KS = NKVn * Dn;  // 1024

  // stage Q tile (64x128) once
#pragma unroll
  for (int c = 0; c < 8; ++c) {
    int lin = tid + c * 128;  // 0..1023
    int row = lin >> 4;
    int col = (lin & 15) << 3;
    *(uint4*)(&Qs[row * 128 + col]) =
        *(const uint4*)(Q + (size_t)(q0 + row) * QS + h * Dn + col);
  }
  __syncthreads();

  float m_r[8], l_r[8];
  v8f o[8];
#pragma unroll
  for (int r = 0; r < 8; ++r) { m_r[r] = -__builtin_inff(); l_r[r] = 0.0f; }
#pragma unroll
  for (int dt = 0; dt < 8; ++dt) o[dt] = zero8();

  for (int kb = 0; kb <= q0; kb += 64) {
    __syncthreads();  // previous iteration's readers of Ks/Vt are done
    // stage K block (64x128) and V block transposed (Vt[d][key])
#pragma unroll
    for (int c = 0; c < 8; ++c) {
      int lin = tid + c * 128;
      int row = lin >> 4;
      int col = (lin & 15) << 3;
      *(uint4*)(&Ks[row * 128 + col]) =
          *(const uint4*)(Kp + (size_t)(kb + row) * KS + kvh * Dn + col);
      uint4 v = *(const uint4*)(Vp + (size_t)(kb + row) * KS + kvh * Dn + col);
      const bf16* pv = (const bf16*)&v;
#pragma unroll
      for (int i = 0; i < 8; ++i) Vt[(col + i) * 64 + row] = pv[i];
    }
    __syncthreads();

    // S = Q * K^T  (contraction over d=128 -> 4 WMMA k-steps; 4 key tiles)
    v8f s[4];
#pragma unroll
    for (int t = 0; t < 4; ++t) s[t] = zero8();
#pragma unroll
    for (int ks = 0; ks < 4; ++ks) {
      v16bf aq = frag_a(&Qs[(wid * 16 + lc) * 128 + ks * 32], half * 8);
#pragma unroll
      for (int t = 0; t < 4; ++t) {
        v16bf bk = frag_b(&Ks[(t * 16 + lc) * 128 + ks * 32 + half * 16]);
        s[t] = wmma_bf16(aq, bk, s[t]);
      }
    }

    // online softmax: mask, row-max / row-sum via shfl within 16-lane halves
    float alpha[8];
#pragma unroll
    for (int r = 0; r < 8; ++r) {
      const int qrow = q0 + wid * 16 + half * 8 + r;
      float mx = -__builtin_inff();
#pragma unroll
      for (int t = 0; t < 4; ++t) {
        int j = kb + t * 16 + lc;
        float sv = s[t][r];
        if (j > qrow) { sv = -__builtin_inff(); s[t][r] = sv; }
        mx = fmaxf(mx, sv);
      }
#pragma unroll
      for (int off = 1; off < 16; off <<= 1)
        mx = fmaxf(mx, __shfl_xor(mx, off, 32));
      float mnew = fmaxf(m_r[r], mx);
      alpha[r] = exp2f((m_r[r] - mnew) * LOG2E);
      float psum = 0.0f;
#pragma unroll
      for (int t = 0; t < 4; ++t) {
        float p = exp2f((s[t][r] - mnew) * LOG2E);
        s[t][r] = p;
        psum += p;
      }
#pragma unroll
      for (int off = 1; off < 16; off <<= 1) psum += __shfl_xor(psum, off, 32);
      l_r[r] = l_r[r] * alpha[r] + psum;
      m_r[r] = mnew;
    }
#pragma unroll
    for (int dt = 0; dt < 8; ++dt)
#pragma unroll
      for (int r = 0; r < 8; ++r) o[dt][r] *= alpha[r];

    // restripe P through per-wave LDS into the A-fragment layout
#pragma unroll
    for (int r = 0; r < 8; ++r)
#pragma unroll
      for (int t = 0; t < 4; ++t)
        Ps[wid * 1024 + (half * 8 + r) * 64 + t * 16 + lc] = (bf16)s[t][r];
    // same-wave DS ops are in order: loads below see the stores above

    // O += P * V  (contraction over 64 keys -> 2 WMMA k-steps; 8 d tiles)
#pragma unroll
    for (int ks2 = 0; ks2 < 2; ++ks2) {
      v16bf ap = frag_a(&Ps[wid * 1024 + lc * 64 + ks2 * 32], half * 8);
#pragma unroll
      for (int dt = 0; dt < 8; ++dt) {
        v16bf bv = frag_b(&Vt[(dt * 16 + lc) * 64 + ks2 * 32 + half * 16]);
        o[dt] = wmma_bf16(ap, bv, o[dt]);
      }
    }
  }

  // epilogue: normalize and store bf16 into [T, NQ*D]
#pragma unroll
  for (int dt = 0; dt < 8; ++dt)
#pragma unroll
    for (int r = 0; r < 8; ++r) {
      int row = q0 + wid * 16 + half * 8 + r;
      int col = h * Dn + dt * 16 + lc;
      O[(size_t)row * QS + col] = (bf16)(o[dt][r] / l_r[r]);
    }
}

// ---------------------------------------------------------------------------
// launcher
// ---------------------------------------------------------------------------
extern "C" void kernel_launch(void* const* d_in, const int* in_sizes, int n_in,
                              void* d_out, int out_size, void* d_ws,
                              size_t ws_size, hipStream_t stream) {
  (void)in_sizes; (void)n_in; (void)out_size; (void)ws_size;
  const float* x    = (const float*)d_in[0];
  const float* cosT = (const float*)d_in[1];
  const float* sinT = (const float*)d_in[2];
  const float* Wq   = (const float*)d_in[3];
  const float* Wk   = (const float*)d_in[4];
  const float* Wv   = (const float*)d_in[5];
  const float* Wo   = (const float*)d_in[6];
  const float* qw   = (const float*)d_in[7];
  const float* kw   = (const float*)d_in[8];
  float* out = (float*)d_out;

  char* p = (char*)d_ws;
  auto alloc = [&](size_t bytes) {
    void* r = (void*)p;
    p += (bytes + 255) & ~(size_t)255;
    return r;
  };
  // ~88 MB total workspace
  bf16* xb  = (bf16*)alloc((size_t)Tn * HIDn * 2);        // x bf16
  bf16* wqb = (bf16*)alloc((size_t)HIDn * NQn * Dn * 2);  // Wq bf16
  bf16* wkb = (bf16*)alloc((size_t)HIDn * NKVn * Dn * 2);
  bf16* wvb = (bf16*)alloc((size_t)HIDn * NKVn * Dn * 2);
  bf16* wob = (bf16*)alloc((size_t)NQn * Dn * HIDn * 2);
  float* qf = (float*)alloc((size_t)Tn * NQn * Dn * 4);   // Q proj f32
  float* kf = (float*)alloc((size_t)Tn * NKVn * Dn * 4);
  float* vf = (float*)alloc((size_t)Tn * NKVn * Dn * 4);
  bf16* qb  = (bf16*)alloc((size_t)Tn * NQn * Dn * 2);    // Q post-norm/rope
  bf16* kb  = (bf16*)alloc((size_t)Tn * NKVn * Dn * 2);
  bf16* vb  = (bf16*)alloc((size_t)Tn * NKVn * Dn * 2);
  bf16* ab  = (bf16*)alloc((size_t)Tn * NQn * Dn * 2);    // attention out

  // 1) converts
  k_cvt_bf16<<<1024, 256, 0, stream>>>(xb, x, Tn * HIDn);
  k_cvt_bf16<<<1024, 256, 0, stream>>>(wqb, Wq, HIDn * NQn * Dn);
  k_cvt_bf16<<<1024, 256, 0, stream>>>(wkb, Wk, HIDn * NKVn * Dn);
  k_cvt_bf16<<<1024, 256, 0, stream>>>(wvb, Wv, HIDn * NKVn * Dn);
  k_cvt_bf16<<<1024, 256, 0, stream>>>(wob, Wo, NQn * Dn * HIDn);

  // 2) projections (WMMA GEMMs, compile-time shapes)
  k_gemm_bf16<Tn, NQn * Dn, HIDn>
      <<<dim3(NQn * Dn / GBN, Tn / GBM), 128, 0, stream>>>(qf, xb, wqb);
  k_gemm_bf16<Tn, NKVn * Dn, HIDn>
      <<<dim3(NKVn * Dn / GBN, Tn / GBM), 128, 0, stream>>>(kf, xb, wkb);
  k_gemm_bf16<Tn, NKVn * Dn, HIDn>
      <<<dim3(NKVn * Dn / GBN, Tn / GBM), 128, 0, stream>>>(vf, xb, wvb);

  // 3) RMSNorm + RoPE (Q carries the D^-0.5 softmax scale), V convert
  const float qscale = 0.08838834764831845f;  // 128^-0.5
  k_rmsrope<<<dim3(Tn, NQn), 128, 0, stream>>>(qb, qf, cosT, sinT, qw, NQn,
                                               qscale);
  k_rmsrope<<<dim3(Tn, NKVn), 128, 0, stream>>>(kb, kf, cosT, sinT, kw, NKVn,
                                                1.0f);
  k_cvt_bf16<<<1024, 256, 0, stream>>>(vb, vf, Tn * NKVn * Dn);

  // 4) causal flash attention (WMMA for QK^T and PV)
  k_attn<<<dim3(Tn / 64, NQn), 128, 0, stream>>>(ab, qb, kb, vb);

  // 5) output projection -> f32 d_out
  k_gemm_bf16<Tn, HIDn, NQn * Dn>
      <<<dim3(HIDn / GBN, Tn / GBM), 128, 0, stream>>>(out, ab, wob);
}
